// Model_rits_att_14714557956469
// MI455X (gfx1250) — compile-verified, matrix operand order checked by reading, not run on hardware
//
#include <hip/hip_runtime.h>
#include <math.h>

// ---------------- problem sizes ----------------
#define B_SZ   4096
#define T_SZ   48
#define N_SZ   35
#define H_SZ   128
#define G_SZ   512          // 4*H
#define ATT_SZ 350
#define HOP_SZ 30
#define N2_SZ  70           // 2*N

// ---------------- WMMA types ----------------
typedef __bf16 bf16_t;
typedef __attribute__((ext_vector_type(16))) __bf16 v16bf;
typedef __attribute__((ext_vector_type(8)))  float  v8f;

__device__ __forceinline__ bf16_t f2bf(float f) {
  unsigned u = __builtin_bit_cast(unsigned, f);
  return __builtin_bit_cast(bf16_t, (unsigned short)(u >> 16));
}
__device__ __forceinline__ unsigned short f2bfbits(float f) {
  return (unsigned short)(__builtin_bit_cast(unsigned, f) >> 16);
}

// ---- fast transcendentals: native TRANS-unit ops (v_exp_f32 / v_rcp_f32) ----
__device__ __forceinline__ float fexp(float x)  { return __expf(x); }
__device__ __forceinline__ float frcp(float x)  { return __builtin_amdgcn_rcpf(x); }
__device__ __forceinline__ float fsigmoid(float x) { return frcp(1.0f + __expf(-x)); }
__device__ __forceinline__ float ftanh(float x) {
  float ax = fabsf(x);
  float e = __expf(-2.0f * ax);
  float r = (1.0f - e) * frcp(1.0f + e);
  return copysignf(r, x);
}

// Async copy: global -> LDS, one dword, tracked by ASYNCcnt (no VGPR transit).
// Generic shared pointers on amdgcn carry the LDS offset in their low 32 bits.
__device__ __forceinline__ void async_b32(void* lds, const void* g) {
  unsigned off = (unsigned)(uintptr_t)lds;
  asm volatile("global_load_async_to_lds_b32 %0, %1, off"
               :: "v"(off), "v"(g) : "memory");
}
__device__ __forceinline__ void wait_async0() {
  asm volatile("s_wait_asynccnt 0x0" ::: "memory");
}

// Build a 16x32 bf16 A-fragment from an fp32 row-major LDS tile (M=16 rows).
// Layout per CDNA5 ISA 7.12.2: lane<16 holds row M=lane, K in {0..7,16..23};
// lane>=16 holds row M=lane-16, K in {8..15,24..31} (each +kt*32).
template <int KVAL>
__device__ __forceinline__ v16bf load_a_f32(const float* X, int ldx, int kt, int lane) {
  const int half = lane >> 4;
  const int m = lane & 15;
  v16bf a;
#pragma unroll
  for (int e = 0; e < 16; ++e) {
    int kl = (e & 7) + ((e >> 3) << 4) + (half << 3);
    int k = kt * 32 + kl;
    float v = (k < KVAL) ? X[m * ldx + k] : 0.0f;
    a[e] = f2bf(v);
  }
  return a;
}

enum { ACT_NONE = 0, ACT_EXPRELU = 1, ACT_TANH = 2 };

// Y[16,NOUT] (+)= X[16,KVAL] @ W^T (+ bias), W pre-packed into bf16 B-fragments.
// A-fragments are hoisted into registers once; the inner loop is B-load + WMMA.
// 4 waves cooperate: wave w owns n-tiles {w, w+4, ...}. Barriers handled by caller.
template <int KVAL, int NOUT, int ACT, bool ACC>
__device__ __forceinline__ void gemm16(const float* X, int ldx,
                                       const unsigned short* wfrag,
                                       const float* bias,
                                       float* Y, int ldy,
                                       int lane, int wave) {
  constexpr int KT = (KVAL + 31) / 32;
  constexpr int NT = (NOUT + 15) / 16;
  if (wave >= NT) return;  // uniform per wave; WMMA waves keep EXEC all-ones
  v16bf afr[KT];
#pragma unroll
  for (int kt = 0; kt < KT; ++kt) afr[kt] = load_a_f32<KVAL>(X, ldx, kt, lane);
#pragma unroll
  for (int nt0 = 0; nt0 < (NT + 3) / 4; ++nt0) {
    const int nt = wave + nt0 * 4;
    if (nt >= NT) break;
    v8f acc;
#pragma unroll
    for (int i = 0; i < 8; ++i) acc[i] = 0.0f;
#pragma unroll
    for (int kt = 0; kt < KT; ++kt) {
      v16bf b = *(const v16bf*)(wfrag + (((long)(nt * KT + kt)) << 9) + (lane << 4));
      acc = __builtin_amdgcn_wmma_f32_16x16x32_bf16(false, afr[kt], false, b, (short)0,
                                                    acc, false, false);
    }
    const int col = (nt << 4) + (lane & 15);
    const int half = lane >> 4;
    if (col < NOUT) {
      float bv = bias[col];
#pragma unroll
      for (int r = 0; r < 8; ++r) {
        int row = r + (half << 3);
        float v = acc[r] + bv;
        if (ACC) v += Y[row * ldy + col];
        if (ACT == ACT_EXPRELU)      v = fexp(-fmaxf(v, 0.0f));
        else if (ACT == ACT_TANH)    v = ftanh(v);
        Y[row * ldy + col] = v;
      }
    }
  }
}

// ---------------- LDS staging (unioned; ~51 KB) ----------------
struct SMem {
  float x[16][N_SZ];
  float m[16][N_SZ];
  float d[16][N_SZ];
  float h[16][H_SZ];          // hidden state, persists across steps
  float cat[16][N2_SZ];       // concat buffers (gamma_x|m, then c_c|m, then s2 out)
  union {
    float gh[16][H_SZ];       // gamma_h
    struct { float xh[16][N_SZ]; float xc[16][N_SZ];
             float zh[16][N_SZ]; float al[16][N_SZ]; } p;
    float gates[16][G_SZ];    // LSTM gates
    float s1[16][ATT_SZ + 2]; // tanh(h @ Ws1^T) (stride padded to 352)
  } u;
};

// ---------------- weight pre-pack into WMMA B-fragment layout ----------------
extern "C" __global__ void pack_weight_kernel(const float* __restrict__ W,
                                              unsigned short* __restrict__ dst,
                                              int Nout, int K, int zero_diag, int total) {
  const int ktiles = (K + 31) >> 5;
  for (int idx = blockIdx.x * blockDim.x + threadIdx.x; idx < total;
       idx += gridDim.x * blockDim.x) {
    int tile = idx >> 9, within = idx & 511;
    int lane = within >> 4, e = within & 15;
    int nt = tile / ktiles, kt = tile - nt * ktiles;
    int half = lane >> 4, lp = lane & 15;
    int n = (nt << 4) + lp;
    int kl = (e & 7) + ((e >> 3) << 4) + (half << 3);
    int k = (kt << 5) + kl;
    float v = 0.0f;
    if (n < Nout && k < K && !(zero_diag && n == k)) v = W[n * K + k];
    dst[idx] = f2bfbits(v);
  }
}

extern "C" __global__ void zero_kernel(float* __restrict__ p, int n) {
  int i = blockIdx.x * blockDim.x + threadIdx.x;
  if (i < n) p[i] = 0.0f;
}

// ---------------- persistent RNN kernel: 16 batch rows per block ----------------
extern "C" __global__ void __launch_bounds__(128)
rits_rnn_kernel(const float* __restrict__ values, const float* __restrict__ masks,
                const float* __restrict__ deltas,
                const float* __restrict__ td_x_W, const float* __restrict__ td_x_b,
                const float* __restrict__ td_h_b, const float* __restrict__ hist_b,
                const float* __restrict__ feat_b, const float* __restrict__ wc_b,
                const float* __restrict__ bih, const float* __restrict__ bhh,
                const float* __restrict__ ws1_b, const float* __restrict__ ws2_b,
                const unsigned short* __restrict__ f_tdh, const unsigned short* __restrict__ f_hist,
                const unsigned short* __restrict__ f_feat, const unsigned short* __restrict__ f_wc,
                const unsigned short* __restrict__ f_wih, const unsigned short* __restrict__ f_whh,
                const unsigned short* __restrict__ f_ws1, const unsigned short* __restrict__ f_ws2,
                unsigned short* __restrict__ Hseq, float* __restrict__ Sseq,
                float* __restrict__ num_acc, float* __restrict__ den_acc,
                float* __restrict__ imput) {
  __shared__ SMem sm;
  const int tid = threadIdx.x, lane = tid & 31, wave = tid >> 5;
  const long b0 = (long)blockIdx.x * 16;

  float creg[16];
#pragma unroll
  for (int r = 0; r < 16; ++r) creg[r] = 0.0f;
  for (int i = tid; i < 16 * H_SZ; i += 128) (&sm.h[0][0])[i] = 0.0f;
  __syncthreads();

  for (int t = 0; t < T_SZ; ++t) {
    // stage x, m, d for this timestep via async global->LDS copies (ASYNCcnt)
    for (int i = tid; i < 16 * N_SZ; i += 128) {
      int r = i / N_SZ, n = i - r * N_SZ;
      long g = ((b0 + r) * T_SZ + t) * N_SZ + n;
      async_b32(&sm.x[r][n], values + g);
      async_b32(&sm.m[r][n], masks + g);
      async_b32(&sm.d[r][n], deltas + g);
      if (n == 0 && t + 1 < T_SZ) {  // prefetch next step's rows into L2
        __builtin_prefetch(values + g + N_SZ, 0, 1);
        __builtin_prefetch(masks + g + N_SZ, 0, 1);
        __builtin_prefetch(deltas + g + N_SZ, 0, 1);
      }
    }
    wait_async0();
    __syncthreads();

    // gamma_h = exp(-relu(d @ td_h_W^T + b))
    gemm16<N_SZ, H_SZ, ACT_EXPRELU, false>(&sm.d[0][0], N_SZ, f_tdh, td_h_b,
                                           &sm.u.gh[0][0], H_SZ, lane, wave);
    __syncthreads();
    for (int i = tid; i < 16 * H_SZ; i += 128) {
      int r = i >> 7, j = i & 127;
      sm.h[r][j] *= sm.u.gh[r][j];
    }
    __syncthreads();

    // x_h = h @ hist_W^T + b
    gemm16<H_SZ, N_SZ, ACT_NONE, false>(&sm.h[0][0], H_SZ, f_hist, hist_b,
                                        &sm.u.p.xh[0][0], N_SZ, lane, wave);
    __syncthreads();

    float p_num = 0.0f, p_den = 0.0f;
    // gamma_x (diag), x_c, alpha input (gamma_x|m), loss term 1
    for (int i = tid; i < 16 * N_SZ; i += 128) {
      int r = i / N_SZ, n = i - r * N_SZ;
      float dd = sm.d[r][n], mm = sm.m[r][n], xx = sm.x[r][n], xh = sm.u.p.xh[r][n];
      float gx = fexp(-fmaxf(dd * td_x_W[n * N_SZ + n] + td_x_b[n], 0.0f));
      sm.cat[r][n] = gx;
      sm.cat[r][N_SZ + n] = mm;
      sm.u.p.xc[r][n] = mm * xx + (1.0f - mm) * xh;
      p_num += fabsf(xx - xh) * mm;
      p_den += mm;
    }
    __syncthreads();

    // z_h = x_c @ feat_W(zero diag)^T + b
    gemm16<N_SZ, N_SZ, ACT_NONE, false>(&sm.u.p.xc[0][0], N_SZ, f_feat, feat_b,
                                        &sm.u.p.zh[0][0], N_SZ, lane, wave);
    __syncthreads();
    // alpha = [gamma_x|m] @ wc_W^T + b
    gemm16<N2_SZ, N_SZ, ACT_NONE, false>(&sm.cat[0][0], N2_SZ, f_wc, wc_b,
                                         &sm.u.p.al[0][0], N_SZ, lane, wave);
    __syncthreads();

    // c_h, c_c, imputations, loss terms 2+3, build [c_c|m]
    for (int i = tid; i < 16 * N_SZ; i += 128) {
      int r = i / N_SZ, n = i - r * N_SZ;
      float mm = sm.m[r][n], xx = sm.x[r][n];
      float xh = sm.u.p.xh[r][n], zh = sm.u.p.zh[r][n], a = sm.u.p.al[r][n];
      float ch = a * zh + (1.0f - a) * xh;
      float cc = mm * xx + (1.0f - mm) * ch;
      p_num += (fabsf(xx - zh) + fabsf(xx - ch)) * mm;
      imput[((b0 + r) * T_SZ + t) * N_SZ + n] = cc;
      sm.cat[r][n] = cc;
      sm.cat[r][N_SZ + n] = mm;
    }
    __syncthreads();

    // gates = [c_c|m] @ Wih^T + bih  +  h @ Whh^T + bhh
    gemm16<N2_SZ, G_SZ, ACT_NONE, false>(&sm.cat[0][0], N2_SZ, f_wih, bih,
                                         &sm.u.gates[0][0], G_SZ, lane, wave);
    __syncthreads();
    gemm16<H_SZ, G_SZ, ACT_NONE, true>(&sm.h[0][0], H_SZ, f_whh, bhh,
                                       &sm.u.gates[0][0], G_SZ, lane, wave);
    __syncthreads();

    // LSTM cell update; thread tid owns hidden column j=tid for all 16 rows
    {
      const int j = tid;
#pragma unroll
      for (int r = 0; r < 16; ++r) {
        float ig = sm.u.gates[r][j],       fg = sm.u.gates[r][128 + j];
        float gg = sm.u.gates[r][256 + j], og = sm.u.gates[r][384 + j];
        float c = fsigmoid(fg) * creg[r] + fsigmoid(ig) * ftanh(gg);
        float hn = fsigmoid(og) * ftanh(c);
        creg[r] = c;
        sm.h[r][j] = hn;
        Hseq[((b0 + r) * T_SZ + t) * H_SZ + j] = f2bfbits(hn);
      }
    }
    __syncthreads();

    // attention scores: s = tanh(h @ Ws1^T + b) @ Ws2^T + b  -> Sseq[b][t][30]
    gemm16<H_SZ, ATT_SZ, ACT_TANH, false>(&sm.h[0][0], H_SZ, f_ws1, ws1_b,
                                          &sm.u.s1[0][0], ATT_SZ + 2, lane, wave);
    __syncthreads();
    gemm16<ATT_SZ, HOP_SZ, ACT_NONE, false>(&sm.u.s1[0][0], ATT_SZ + 2, f_ws2, ws2_b,
                                            &sm.cat[0][0], N2_SZ, lane, wave);
    __syncthreads();
    for (int i = tid; i < 16 * HOP_SZ; i += 128) {
      int r = i / HOP_SZ, k = i - r * HOP_SZ;
      Sseq[((b0 + r) * T_SZ + t) * HOP_SZ + k] = sm.cat[r][k];
    }

    // per-step loss partials: wave32 shuffle-reduce, one atomic per wave
#pragma unroll
    for (int off = 16; off > 0; off >>= 1) {
      p_num += __shfl_xor(p_num, off, 32);
      p_den += __shfl_xor(p_den, off, 32);
    }
    if (lane == 0) {
      atomicAdd(&num_acc[t], p_num);
      atomicAdd(&den_acc[t], p_den);
    }
    __syncthreads();
  }
}

// ------------- attention pooling + output head: y = sum_{t,k} attn[k,t]*P[t,k] -------------
extern "C" __global__ void __launch_bounds__(128)
rits_attn_kernel(const unsigned short* __restrict__ Hseq, const float* __restrict__ Sseq,
                 const unsigned short* __restrict__ f_wo, const float* __restrict__ out_b,
                 const float* __restrict__ labels, const float* __restrict__ is_train,
                 float* __restrict__ preds, float* __restrict__ yacc) {
  __shared__ float P[4][T_SZ][HOP_SZ];  // per-wave slab: P = H_b @ Wo^T  [48 x 30]
  const int tid = threadIdx.x, lane = tid & 31, wave = tid >> 5;
  const long b = (long)blockIdx.x * 4 + wave;
  const int half = lane >> 4, ml = lane & 15;

#pragma unroll
  for (int mt = 0; mt < 3; ++mt) {        // M = t (48 = 3*16)
    // hoist A-fragments (rows t = mt*16.., K = H) once per m-tile
    v16bf afr[4];
#pragma unroll
    for (int kt = 0; kt < 4; ++kt) {
      v16bf a;
#pragma unroll
      for (int e = 0; e < 16; ++e) {
        int kl = (e & 7) + ((e >> 3) << 4) + (half << 3);
        int tt = mt * 16 + ml;
        a[e] = __builtin_bit_cast(bf16_t, Hseq[(b * T_SZ + tt) * H_SZ + kt * 32 + kl]);
      }
      afr[kt] = a;
    }
#pragma unroll
    for (int nt = 0; nt < 2; ++nt) {      // N = hop (30 -> 2 tiles)
      v8f acc;
#pragma unroll
      for (int i = 0; i < 8; ++i) acc[i] = 0.0f;
#pragma unroll
      for (int kt = 0; kt < 4; ++kt) {    // K = H (128)
        v16bf bfrag = *(const v16bf*)(f_wo + ((nt * 4 + kt) << 9) + (lane << 4));
        acc = __builtin_amdgcn_wmma_f32_16x16x32_bf16(false, afr[kt], false, bfrag,
                                                      (short)0, acc, false, false);
      }
      int col = (nt << 4) + ml;
      if (col < HOP_SZ) {
#pragma unroll
        for (int r = 0; r < 8; ++r)
          P[wave][mt * 16 + r + (half << 3)][col] = acc[r];
      }
    }
  }
  __syncthreads();

  // lane k < 30: softmax over T for hop k, then sum attn * P
  float res = 0.0f;
  if (lane < HOP_SZ) {
    float mx = -3.0e38f;
    for (int t = 0; t < T_SZ; ++t)
      mx = fmaxf(mx, Sseq[(b * T_SZ + t) * HOP_SZ + lane]);
    float Z = 0.0f;
    for (int t = 0; t < T_SZ; ++t)
      Z += fexp(Sseq[(b * T_SZ + t) * HOP_SZ + lane] - mx);
    float a = 0.0f;
    for (int t = 0; t < T_SZ; ++t)
      a += fexp(Sseq[(b * T_SZ + t) * HOP_SZ + lane] - mx) * P[wave][t][lane];
    res = a * frcp(Z);
  }
#pragma unroll
  for (int off = 16; off > 0; off >>= 1) res += __shfl_xor(res, off, 32);

  if (lane == 0) {
    float y = res + out_b[0];
    preds[b] = fsigmoid(y);
    float lab = labels[b], it = is_train[b];
    float mv = fmaxf(-y, 0.0f);
    float bce = y - y * lab + mv + __logf(__expf(-mv) + __expf(-y - mv));
    atomicAdd(&yacc[0], bce * it);
    atomicAdd(&yacc[1], it);
  }
}

extern "C" __global__ void finalize_kernel(const float* __restrict__ num,
                                           const float* __restrict__ den,
                                           const float* __restrict__ yacc,
                                           float* __restrict__ out0) {
  if (threadIdx.x == 0 && blockIdx.x == 0) {
    float xl = 0.0f;
    for (int t = 0; t < T_SZ; ++t) xl += num[t] / (den[t] + 1e-5f);
    float yl = yacc[0] / (yacc[1] + 1e-5f);
    out0[0] = xl / (float)T_SZ + 0.1f * yl;
  }
}

// ---------------- host-side orchestration ----------------
extern "C" void kernel_launch(void* const* d_in, const int* in_sizes, int n_in,
                              void* d_out, int out_size, void* d_ws, size_t ws_size,
                              hipStream_t stream) {
  (void)in_sizes; (void)n_in; (void)out_size; (void)ws_size;
  const float* values   = (const float*)d_in[0];
  const float* masks    = (const float*)d_in[1];
  const float* deltas   = (const float*)d_in[2];
  const float* labels   = (const float*)d_in[3];
  const float* is_train = (const float*)d_in[4];
  const float* td_h_W   = (const float*)d_in[5];
  const float* td_h_b   = (const float*)d_in[6];
  const float* td_x_W   = (const float*)d_in[7];
  const float* td_x_b   = (const float*)d_in[8];
  const float* hist_W   = (const float*)d_in[9];
  const float* hist_b   = (const float*)d_in[10];
  const float* feat_W   = (const float*)d_in[11];
  const float* feat_b   = (const float*)d_in[12];
  const float* wc_W     = (const float*)d_in[13];
  const float* wc_b     = (const float*)d_in[14];
  const float* Wih      = (const float*)d_in[15];
  const float* bih      = (const float*)d_in[16];
  const float* Whh      = (const float*)d_in[17];
  const float* bhh      = (const float*)d_in[18];
  const float* Ws1_W    = (const float*)d_in[19];
  const float* Ws1_b    = (const float*)d_in[20];
  const float* Ws2_W    = (const float*)d_in[21];
  const float* Ws2_b    = (const float*)d_in[22];
  const float* out_W    = (const float*)d_in[23];
  const float* out_b    = (const float*)d_in[24];

  auto ntf = [](int N) { return (N + 15) / 16; };
  auto ktf = [](int K) { return (K + 31) / 32; };

  unsigned short* p = (unsigned short*)d_ws;
  auto carve = [&](int Nout, int K) {
    unsigned short* q = p;
    p += (size_t)ntf(Nout) * ktf(K) * 512;
    return q;
  };
  unsigned short* f_tdh  = carve(H_SZ, N_SZ);
  unsigned short* f_hist = carve(N_SZ, H_SZ);
  unsigned short* f_feat = carve(N_SZ, N_SZ);
  unsigned short* f_wc   = carve(N_SZ, N2_SZ);
  unsigned short* f_wih  = carve(G_SZ, N2_SZ);
  unsigned short* f_whh  = carve(G_SZ, H_SZ);
  unsigned short* f_ws1  = carve(ATT_SZ, H_SZ);
  unsigned short* f_ws2  = carve(HOP_SZ, ATT_SZ);
  unsigned short* f_wo   = carve(HOP_SZ, H_SZ);
  unsigned short* Hseq = p;
  p += (size_t)B_SZ * T_SZ * H_SZ;
  float* fp = (float*)p;
  float* Sseq = fp;  fp += (size_t)B_SZ * T_SZ * HOP_SZ;
  float* numA = fp;  fp += T_SZ;
  float* denA = fp;  fp += T_SZ;
  float* yacc = fp;  fp += 2;

  // zero per-step loss accumulators (numA, denA, yacc are contiguous)
  zero_kernel<<<1, 128, 0, stream>>>(numA, T_SZ * 2 + 2);

  auto pack = [&](const float* W, unsigned short* dst, int Nout, int K, int zd) {
    int total = ntf(Nout) * ktf(K) * 512;
    pack_weight_kernel<<<(total + 255) / 256, 256, 0, stream>>>(W, dst, Nout, K, zd, total);
  };
  pack(td_h_W, f_tdh, H_SZ, N_SZ, 0);
  pack(hist_W, f_hist, N_SZ, H_SZ, 0);
  pack(feat_W, f_feat, N_SZ, N_SZ, 1);   // zero-diag feature regression
  pack(wc_W,   f_wc,   N_SZ, N2_SZ, 0);
  pack(Wih,    f_wih,  G_SZ, N2_SZ, 0);
  pack(Whh,    f_whh,  G_SZ, H_SZ, 0);
  pack(Ws1_W,  f_ws1,  ATT_SZ, H_SZ, 0);
  pack(Ws2_W,  f_ws2,  HOP_SZ, ATT_SZ, 0);
  pack(out_W,  f_wo,   HOP_SZ, H_SZ, 0); // out_W reshaped [30][128]

  float* out   = (float*)d_out;
  float* preds = out + 1;
  float* imput = out + 1 + B_SZ;

  rits_rnn_kernel<<<B_SZ / 16, 128, 0, stream>>>(
      values, masks, deltas, td_x_W, td_x_b, td_h_b, hist_b, feat_b, wc_b,
      bih, bhh, Ws1_b, Ws2_b,
      f_tdh, f_hist, f_feat, f_wc, f_wih, f_whh, f_ws1, f_ws2,
      Hseq, Sseq, numA, denA, imput);

  rits_attn_kernel<<<B_SZ / 4, 128, 0, stream>>>(
      Hseq, Sseq, f_wo, out_b, labels, is_train, preds, yacc);

  finalize_kernel<<<1, 32, 0, stream>>>(numA, denA, yacc, out);
}
